// CausalSelfAttention_2740189134905
// MI455X (gfx1250) — compile-verified, matrix operand order checked by reading, not run on hardware
//
#include <hip/hip_runtime.h>
#include <hip/hip_bf16.h>

typedef __attribute__((ext_vector_type(16))) _Float16 v16h;
typedef __attribute__((ext_vector_type(8)))  _Float16 v8h;
typedef __attribute__((ext_vector_type(8)))  float    v8f;
typedef __attribute__((ext_vector_type(4)))  unsigned int v4u;
typedef __attribute__((ext_vector_type(8)))  int v8i;
typedef __attribute__((ext_vector_type(4)))  int v4i;

constexpr int SEQ   = 2048;
constexpr int NHEAD = 16;
constexpr int NKV   = 4;
constexpr int HDIM  = 128;
constexpr int EMB   = 2048;   // NHEAD * HDIM
constexpr int KVD   = 512;    // NKV * HDIM
constexpr int HPKC  = 4;      // NHEAD / NKV
constexpr int GIN   = 12;

// ---------------------------------------------------------------------------
// WMMA fragment helpers (CDNA5 16x16x32 f16 layout, wave32)
// A/B 16x32: lanes 0-15 -> row=lane,  K=[0..7]&[16..23]
//            lanes 16-31 -> row=lane-16, K=[8..15]&[24..31]
// C/D 16x16: VGPR r: lanes 0-15 -> M=r, N=lane; lanes 16-31 -> M=r+8, N=lane-16
// ---------------------------------------------------------------------------
__device__ __forceinline__ v16h make_frag(v8h lo, v8h hi) {
  v16h f;
#pragma unroll
  for (int i = 0; i < 8; ++i) { f[i] = lo[i]; f[8 + i] = hi[i]; }
  return f;
}

__device__ __forceinline__ v16h load_frag(const _Float16* base, int ld) {
  int lane = threadIdx.x & 31;
  int row  = lane & 15;
  int hl   = lane >> 4;
  const _Float16* p = base + (size_t)row * ld + hl * 8;
  v8h lo = *(const v8h*)(p);
  v8h hi = *(const v8h*)(p + 16);
  return make_frag(lo, hi);
}

__device__ __forceinline__ v8f wmma_f16(v16h a, v16h b, v8f c) {
  return __builtin_amdgcn_wmma_f32_16x16x32_f16(false, a, false, b, (short)0, c,
                                                false, false);
}

__device__ __forceinline__ float redsum32(float v) {
#pragma unroll
  for (int o = 16; o > 0; o >>= 1) v += __shfl_xor(v, o, 32);
  return v;
}
__device__ __forceinline__ float redsum16(float v) {
#pragma unroll
  for (int o = 8; o > 0; o >>= 1) v += __shfl_xor(v, o, 16);
  return v;
}
__device__ __forceinline__ float redmax16(float v) {
#pragma unroll
  for (int o = 8; o > 0; o >>= 1) v = fmaxf(v, __shfl_xor(v, o, 16));
  return v;
}

// ---------------------------------------------------------------------------
// Tensor Data Mover: async DMA of a 2D f16 tile (tile_w contiguous elements x
// tile_h rows, row stride in elements) from global into LDS at lds_off.
// D# per CDNA5 ISA. Tracked by TENSORcnt.
// ---------------------------------------------------------------------------
__device__ __forceinline__ void tdm_load_tile_f16(unsigned lds_off,
                                                  const _Float16* gptr,
                                                  unsigned tile_w, unsigned tile_h,
                                                  unsigned long long row_stride) {
  unsigned long long ga = (unsigned long long)(size_t)gptr;
  v4u g0 = {};
  g0[0] = 1u;                                  // count=1 (user mode, no gather)
  g0[1] = lds_off;                             // lds_addr (bytes)
  g0[2] = (unsigned)ga;                        // global_addr[31:0]
  g0[3] = (unsigned)((ga >> 32) & 0x01ffffffu) | (2u << 30);  // addr[56:32], type=2
  v8i g1 = {};
  g1[0] = (int)(1u << 16);                     // data_size=1 -> 2-byte elements
  g1[1] = (int)((tile_w & 0xffffu) << 16);     // tensor_dim0[15:0]
  g1[2] = (int)(((tile_w >> 16) & 0xffffu) |   // tensor_dim0[31:16]
                ((tile_h & 0xffffu) << 16));   // tensor_dim1[15:0]
  g1[3] = (int)(((tile_h >> 16) & 0xffffu) |   // tensor_dim1[31:16]
                ((tile_w & 0xffffu) << 16));   // tile_dim0
  g1[4] = (int)(tile_h & 0xffffu);             // tile_dim1 (tile_dim2=0)
  g1[5] = (int)(row_stride & 0xffffffffu);     // tensor_dim0_stride[31:0]
  g1[6] = (int)((row_stride >> 32) & 0xffffu); // tensor_dim0_stride[47:32]
  g1[7] = 0;
  v4i gz = {};
#if defined(__clang_major__) && (__clang_major__ >= 23)
  v8i gz8 = {};
  __builtin_amdgcn_tensor_load_to_lds(g0, g1, gz, gz, gz8, 0);
#else
  __builtin_amdgcn_tensor_load_to_lds(g0, g1, gz, gz, 0);
#endif
}

// ---------------------------------------------------------------------------
// fp32 -> fp16 conversion
// ---------------------------------------------------------------------------
__global__ void cvt_f32_to_f16(const float* __restrict__ src,
                               _Float16* __restrict__ dst, int n) {
  int i = blockIdx.x * blockDim.x + threadIdx.x;
  if (i < n) dst[i] = (_Float16)src[i];
}

// ---------------------------------------------------------------------------
// C[M,N] = A[M,K] * B[N,K]^T  (f16 in, f32 out).
// Block = 4 waves -> 64x64 output tile. B tiles (64 rows x 32 k) are DMA'd by
// the TDM into double-buffered LDS (issued by wave 0, overlapped with WMMA);
// A fragments are register-prefetched one K-step ahead.
// ---------------------------------------------------------------------------
__global__ void __launch_bounds__(128)
gemm_wmma(const _Float16* __restrict__ A, const _Float16* __restrict__ B,
          float* __restrict__ C, int M, int N, int K) {
  __shared__ _Float16 Bl[2][64 * 32];
  int nbase = blockIdx.x * 64;
  int mbase = blockIdx.y * 64;
  int wave  = threadIdx.x >> 5;
  int lane  = threadIdx.x & 31;
  int mrow  = mbase + wave * 16;

  if (wave == 0)
    tdm_load_tile_f16((unsigned)(size_t)&Bl[0][0],
                      B + (size_t)nbase * K, 32, 64, (unsigned long long)K);

  v8f acc[4] = {};
  v16h a_cur = load_frag(A + (size_t)mrow * K, K);

  for (int k = 0; k < K; k += 32) {
    bool has_next = (k + 32) < K;
    int buf = (k >> 5) & 1;
    if (wave == 0 && has_next)
      tdm_load_tile_f16((unsigned)(size_t)&Bl[buf ^ 1][0],
                        B + (size_t)nbase * K + (k + 32), 32, 64,
                        (unsigned long long)K);
    if (wave == 0) {
      if (has_next) __builtin_amdgcn_s_wait_tensorcnt(1);
      else          __builtin_amdgcn_s_wait_tensorcnt(0);
    }
    __syncthreads();            // current B tile resident in LDS for all waves
    v16h a_nxt = a_cur;
    if (has_next) a_nxt = load_frag(A + (size_t)mrow * K + (k + 32), K);
    const _Float16* bl = &Bl[buf][0];
#pragma unroll
    for (int j = 0; j < 4; ++j) {
      v16h b = load_frag(bl + (size_t)(j * 16) * 32, 32);
      acc[j] = wmma_f16(a_cur, b, acc[j]);
    }
    a_cur = a_nxt;
    __syncthreads();            // all reads done before next TDM overwrite
  }

  int row0 = mrow + (lane >> 4) * 8;
  int col  = lane & 15;
#pragma unroll
  for (int j = 0; j < 4; ++j)
#pragma unroll
    for (int r = 0; r < 8; ++r)
      C[(size_t)(row0 + r) * N + nbase + j * 16 + col] = acc[j][r];
}

// ---------------------------------------------------------------------------
// Per-head RMSNorm + RoPE + optional gain + cross-head shared mean of last 16
// dims. One wave per head; one block per sequence position.
// ---------------------------------------------------------------------------
__global__ void posthead(const float* __restrict__ in, _Float16* __restrict__ outh,
                         const float* __restrict__ cosb,
                         const float* __restrict__ sinb,
                         const float* __restrict__ gain, int nh) {
  int sidx = blockIdx.x;
  int wave = threadIdx.x >> 5;
  int lane = threadIdx.x & 31;
  const float* base = in + (size_t)sidx * nh * HDIM + wave * HDIM;
  float v0 = base[lane], v1 = base[lane + 32], v2 = base[lane + 64],
        v3 = base[lane + 96];
  float ss = redsum32(v0 * v0 + v1 * v1 + v2 * v2 + v3 * v3);
  float r = rsqrtf(ss * (1.0f / HDIM) + 1.1920929e-07f);
  v0 *= r; v1 *= r; v2 *= r; v3 *= r;
  float c0 = cosb[sidx * 64 + lane], s0 = sinb[sidx * 64 + lane];
  float c1 = cosb[sidx * 64 + lane + 32], s1 = sinb[sidx * 64 + lane + 32];
  float n0 =  v0 * c0 + v2 * s0;
  float n2 = -v0 * s0 + v2 * c0;
  float n1 =  v1 * c1 + v3 * s1;
  float n3 = -v1 * s1 + v3 * c1;
  if (gain) { float g = gain[wave]; n0 *= g; n1 *= g; n2 *= g; n3 *= g; }
  __shared__ float sh[16][16];
  if (lane >= 16) sh[wave][lane - 16] = n3;
  __syncthreads();
  if (lane >= 16) {
    float m = 0.f;
    for (int i = 0; i < nh; ++i) m += sh[i][lane - 16];
    n3 = m / nh;
  }
  _Float16* ob = outh + (size_t)sidx * nh * HDIM + wave * HDIM;
  ob[lane]      = (_Float16)n0;
  ob[lane + 32] = (_Float16)n1;
  ob[lane + 64] = (_Float16)n2;
  ob[lane + 96] = (_Float16)n3;
}

// ---------------------------------------------------------------------------
// V epilogue: copy raw fp32 v to output region + build f16 transposed copy
// vt[kv*HDIM + d][seq] so attention PV B-fragments are contiguous loads.
// ---------------------------------------------------------------------------
__global__ void transpose_v(const float* __restrict__ vf, float* __restrict__ vout,
                            _Float16* __restrict__ vt) {
  __shared__ float tile[32][33];
  int s0 = blockIdx.x * 32;
  int c0 = blockIdx.y * 32;
  int tx = threadIdx.x, ty = threadIdx.y;
#pragma unroll
  for (int j = 0; j < 4; ++j) {
    int s = s0 + ty + j * 8;
    float val = vf[(size_t)s * KVD + c0 + tx];
    vout[(size_t)s * KVD + c0 + tx] = val;
    tile[ty + j * 8][tx] = val;
  }
  __syncthreads();
#pragma unroll
  for (int j = 0; j < 4; ++j) {
    int c = c0 + ty + j * 8;
    vt[(size_t)c * SEQ + s0 + tx] = (_Float16)tile[tx][ty + j * 8];
  }
}

// ---------------------------------------------------------------------------
// Flash attention: one wave per (16-query tile, head). Online softmax,
// QK^T and PV both via v_wmma_f32_16x16x32_f16. V read from the transposed
// f16 copy so the PV B-fragment is two contiguous b128 loads per tile.
// ---------------------------------------------------------------------------
__global__ void __launch_bounds__(32)
attn_kernel(const _Float16* __restrict__ qh, const _Float16* __restrict__ kh,
            const _Float16* __restrict__ vt, float* __restrict__ yf) {
  int qt = blockIdx.x;
  int h  = blockIdx.y;
  int kv = h >> 2;  // HPKC == 4
  int lane = threadIdx.x & 31;
  int hl   = lane >> 4;
  int col  = lane & 15;
  int qrow0 = qt * 16;

  __shared__ _Float16 Plds[16 * 32];

  v16h aq[4];
#pragma unroll
  for (int c = 0; c < 4; ++c)
    aq[c] = load_frag(qh + (size_t)qrow0 * EMB + h * HDIM + c * 32, EMB);

  v8f o[8] = {};
  float m[8], l[8];
#pragma unroll
  for (int r = 0; r < 8; ++r) { m[r] = -1e30f; l[r] = 0.f; }

  const float scale = 0.08838834764831845f;  // 1/sqrt(128)
  int kend = qrow0 + 16;
  for (int kt = 0; kt < kend; kt += 32) {
    v8f s0 = {}, s1 = {};
#pragma unroll
    for (int c = 0; c < 4; ++c) {
      v16h b0 = load_frag(kh + (size_t)kt * KVD + kv * HDIM + c * 32, KVD);
      v16h b1 = load_frag(kh + (size_t)(kt + 16) * KVD + kv * HDIM + c * 32, KVD);
      s0 = wmma_f16(aq[c], b0, s0);
      s1 = wmma_f16(aq[c], b1, s1);
    }
    float alpha[8];
#pragma unroll
    for (int r = 0; r < 8; ++r) {
      int row = qrow0 + r + 8 * hl;
      float x0 = s0[r] * scale;
      float x1 = s1[r] * scale;
      if (kt + col > row)      x0 = -1e30f;
      if (kt + 16 + col > row) x1 = -1e30f;
      float tm = redmax16(fmaxf(x0, x1));
      float mn = fmaxf(m[r], tm);
      alpha[r] = __expf(m[r] - mn);
      float p0 = __expf(x0 - mn);
      float p1 = __expf(x1 - mn);
      l[r] = l[r] * alpha[r] + redsum16(p0 + p1);
      m[r] = mn;
      int prow = r + 8 * hl;
      Plds[prow * 32 + col]      = (_Float16)p0;
      Plds[prow * 32 + 16 + col] = (_Float16)p1;
    }
    __syncthreads();
    v16h pa = load_frag(Plds, 32);
#pragma unroll
    for (int t = 0; t < 8; ++t) {
#pragma unroll
      for (int r = 0; r < 8; ++r) o[t][r] *= alpha[r];
      v16h vb = load_frag(vt + (size_t)(kv * HDIM + t * 16) * SEQ + kt, SEQ);
      o[t] = wmma_f16(pa, vb, o[t]);
    }
    __syncthreads();
  }
#pragma unroll
  for (int t = 0; t < 8; ++t)
#pragma unroll
    for (int r = 0; r < 8; ++r) {
      int row = qrow0 + r + 8 * hl;
      yf[(size_t)row * EMB + h * HDIM + t * 16 + col] = o[t][r] / l[r];
    }
}

// ---------------------------------------------------------------------------
// Subtract projection onto normalized v, apply sigmoid gate, emit f16 y.
// ---------------------------------------------------------------------------
__global__ void orthogate(const float* __restrict__ yf, const float* __restrict__ vf,
                          const float* __restrict__ x, const float* __restrict__ Wg,
                          _Float16* __restrict__ ygh) {
  int sidx = blockIdx.x;
  int kv   = blockIdx.y;
  int j    = threadIdx.x >> 5;
  int lane = threadIdx.x & 31;
  int h    = kv * HPKC + j;
  const float* vrow = vf + (size_t)sidx * KVD + kv * HDIM;
  float w0 = vrow[lane], w1 = vrow[lane + 32], w2 = vrow[lane + 64],
        w3 = vrow[lane + 96];
  float nrm = sqrtf(redsum32(w0 * w0 + w1 * w1 + w2 * w2 + w3 * w3));
  float inv = 1.0f / fmaxf(nrm, 1e-12f);
  w0 *= inv; w1 *= inv; w2 *= inv; w3 *= inv;
  const float* yrow = yf + (size_t)sidx * EMB + h * HDIM;
  float y0 = yrow[lane], y1 = yrow[lane + 32], y2 = yrow[lane + 64],
        y3 = yrow[lane + 96];
  float proj = redsum32(y0 * w0 + y1 * w1 + y2 * w2 + y3 * w3);
  y0 -= proj * w0; y1 -= proj * w1; y2 -= proj * w2; y3 -= proj * w3;
  float t = 0.f;
  const float* xr = x + (size_t)sidx * EMB;
#pragma unroll
  for (int i = 0; i < GIN; ++i) t += xr[i] * Wg[h * GIN + i];
  float g = 2.0f / (1.0f + __expf(-t));
  _Float16* ob = ygh + (size_t)sidx * EMB + h * HDIM;
  ob[lane]      = (_Float16)(y0 * g);
  ob[lane + 32] = (_Float16)(y1 * g);
  ob[lane + 64] = (_Float16)(y2 * g);
  ob[lane + 96] = (_Float16)(y3 * g);
}

// ---------------------------------------------------------------------------
extern "C" void kernel_launch(void* const* d_in, const int* in_sizes, int n_in,
                              void* d_out, int out_size, void* d_ws, size_t ws_size,
                              hipStream_t stream) {
  (void)in_sizes; (void)n_in; (void)out_size; (void)ws_size;
  const float* x    = (const float*)d_in[0];
  const float* cosb = (const float*)d_in[1];
  const float* sinb = (const float*)d_in[2];
  const float* Wq   = (const float*)d_in[3];
  const float* Wk   = (const float*)d_in[4];
  const float* Wv   = (const float*)d_in[5];
  const float* Wp   = (const float*)d_in[6];
  const float* qg   = (const float*)d_in[7];
  const float* Wg   = (const float*)d_in[8];
  float* out  = (float*)d_out;
  float* vout = out + (size_t)SEQ * EMB;

  char* w = (char*)d_ws;
  auto alloc = [&](size_t bytes) {
    void* p = (void*)w;
    w += (bytes + 255) & ~(size_t)255;
    return p;
  };
  _Float16* xh  = (_Float16*)alloc((size_t)SEQ * EMB * 2);
  _Float16* wqh = (_Float16*)alloc((size_t)EMB * EMB * 2);
  _Float16* wkh = (_Float16*)alloc((size_t)KVD * EMB * 2);
  _Float16* wvh = (_Float16*)alloc((size_t)KVD * EMB * 2);
  _Float16* wph = (_Float16*)alloc((size_t)EMB * EMB * 2);
  float*    qf  = (float*)alloc((size_t)SEQ * EMB * 4);
  float*    kf  = (float*)alloc((size_t)SEQ * KVD * 4);
  float*    vf  = (float*)alloc((size_t)SEQ * KVD * 4);
  _Float16* qhp = (_Float16*)alloc((size_t)SEQ * EMB * 2);
  _Float16* khp = (_Float16*)alloc((size_t)SEQ * KVD * 2);
  _Float16* vt  = (_Float16*)alloc((size_t)KVD * SEQ * 2);
  float*    yf  = (float*)alloc((size_t)SEQ * EMB * 4);
  _Float16* ygh = (_Float16*)alloc((size_t)SEQ * EMB * 2);

  auto cvt = [&](const float* s, _Float16* d, int n) {
    cvt_f32_to_f16<<<(n + 255) / 256, 256, 0, stream>>>(s, d, n);
  };
  cvt(x, xh, SEQ * EMB);
  cvt(Wq, wqh, EMB * EMB);
  cvt(Wk, wkh, KVD * EMB);
  cvt(Wv, wvh, KVD * EMB);
  cvt(Wp, wph, EMB * EMB);

  gemm_wmma<<<dim3(EMB / 64, SEQ / 64), 128, 0, stream>>>(xh, wqh, qf, SEQ, EMB, EMB);
  gemm_wmma<<<dim3(KVD / 64, SEQ / 64), 128, 0, stream>>>(xh, wkh, kf, SEQ, KVD, EMB);
  gemm_wmma<<<dim3(KVD / 64, SEQ / 64), 128, 0, stream>>>(xh, wvh, vf, SEQ, KVD, EMB);

  transpose_v<<<dim3(SEQ / 32, KVD / 32), dim3(32, 8), 0, stream>>>(vf, vout, vt);
  posthead<<<SEQ, NHEAD * 32, 0, stream>>>(qf, qhp, cosb, sinb, qg, NHEAD);
  posthead<<<SEQ, NKV * 32, 0, stream>>>(kf, khp, cosb, sinb, nullptr, NKV);

  attn_kernel<<<dim3(SEQ / 16, NHEAD), 32, 0, stream>>>(qhp, khp, vt, yf);

  orthogate<<<dim3(SEQ, NKV), 128, 0, stream>>>(yf, vf, x, Wg, ygh);

  gemm_wmma<<<dim3(EMB / 64, SEQ / 64), 128, 0, stream>>>(ygh, wph, out, SEQ, EMB, EMB);
}
